// ErrorMemoryBank_47562467836573
// MI455X (gfx1250) — matrix operand order repaired and checked by reading, not compiled, not run on hardware
//
#include <hip/hip_runtime.h>
#include <hip/hip_bf16.h>
#include <math.h>

typedef __attribute__((ext_vector_type(16))) _Float16 v16h;
typedef __attribute__((ext_vector_type(8)))  float    v8f;

// Problem constants (from reference setup_inputs)
#define BATCH   256
#define FEAT    128
#define CAP     100000
#define TOPK    5
#define VROW    168          // HORIZON * NUM_FEATURES = 24*7
#define GROUPS  6250         // CAP / 16  (exact)
#define CHUNK_G 64           // 16-key groups per chunk -> 1024 keys
#define NCHUNK  98           // ceil(GROUPS / CHUNK_G)
#define NEG_INF (-3.402823466e38f)

// ---------------------------------------------------------------------------
// Kernel 1: normalize queries -> f16 in A-fragment layout
//   A (16x32 f16, wave32): lane L holds row M=L%16; element e -> K =
//   kc*32 + (e<8?0:16) + (L/16)*8 + (e&7)     (ISA 7.12.2 16-bit A layout)
// ws_qn[((qt*4+kc)*32 + lane)*16 + e]
// ---------------------------------------------------------------------------
__global__ void prep_q_kernel(const float* __restrict__ query,
                              _Float16* __restrict__ qn) {
  __shared__ float invn[16];
  const int qt  = blockIdx.x;          // 16 tiles of 16 queries
  const int tid = threadIdx.x;         // 64 threads
  if (tid < 16) {
    const float* row = query + (qt * 16 + tid) * FEAT;
    float ss = 0.f;
    for (int f = 0; f < FEAT; ++f) { float x = row[f]; ss += x * x; }
    invn[tid] = 1.0f / fmaxf(sqrtf(ss), 1e-12f);
  }
  __syncthreads();
  if (tid < 32) {
    const int lane = tid;
    const float* row = query + (qt * 16 + (lane & 15)) * FEAT;
    const float sc = invn[lane & 15];
    for (int kc = 0; kc < 4; ++kc) {
      v16h frag;
#pragma unroll
      for (int e = 0; e < 16; ++e) {
        int K = kc * 32 + ((e < 8) ? 0 : 16) + (lane >> 4) * 8 + (e & 7);
        frag[e] = (_Float16)(row[K] * sc);
      }
      *(v16h*)(qn + ((size_t)(qt * 4 + kc) * 32 + lane) * 16) = frag;
    }
  }
}

// ---------------------------------------------------------------------------
// Kernel 2: normalize keys -> f16, stored in B-fragment layout per 16-key group:
//   ws_kn[g*2048 + kc*512 + L*16 + e]  (key = g*16 + L%16, feature K as in the A layout)
// B fragment (32x16): lane L = column (key L%16), elements = 16 K (features, same order as A)
// -> one contiguous 32B load per lane.
// ---------------------------------------------------------------------------
__global__ void prep_k_kernel(const float* __restrict__ keys,
                              _Float16* __restrict__ kn) {
  __shared__ float invn[16];
  const int g   = blockIdx.x;          // 6250 groups
  const int tid = threadIdx.x;         // 64 threads
  if (tid < 16) {
    const float* row = keys + (size_t)(g * 16 + tid) * FEAT;
    float ss = 0.f;
    for (int f = 0; f < FEAT; ++f) { float x = row[f]; ss += x * x; }
    invn[tid] = 1.0f / fmaxf(sqrtf(ss), 1e-12f);
  }
  __syncthreads();
  for (int idx = tid; idx < 2048; idx += 64) {
    int e = idx & 15, ln = (idx >> 4) & 31, kl = ln & 15;
    int f = (idx >> 9) * 32 + ((e < 8) ? 0 : 16) + (ln >> 4) * 8 + (e & 7);
    kn[(size_t)g * 2048 + idx] =
        (_Float16)(keys[(size_t)(g * 16 + kl) * FEAT + f] * invn[kl]);
  }
}

// ---------------------------------------------------------------------------
// Kernel 3: decay[c] = 0.995 ^ (global_step - ts[c])
// ---------------------------------------------------------------------------
__global__ void decay_kernel(const int* __restrict__ ts,
                             const int* __restrict__ gstep,
                             float* __restrict__ decay) {
  int c = blockIdx.x * blockDim.x + threadIdx.x;
  if (c < CAP) {
    float age = (float)(gstep[0] - ts[c]);
    decay[c] = exp2f(age * -0.0072315693f);  // log2(0.995)
  }
}

// ---------------------------------------------------------------------------
// Kernel 4: WMMA GEMM + fused decay + per-chunk top-5
// grid = (NCHUNK, 2); block = 256 (8 waves). Wave w -> query tile qt.
// ---------------------------------------------------------------------------
__global__ __launch_bounds__(256)
void sim_topk_kernel(const _Float16* __restrict__ kn,
                     const _Float16* __restrict__ qn,
                     const float* __restrict__ decay,
                     float* __restrict__ pv, int* __restrict__ pi) {
  const int lane  = threadIdx.x & 31;
  const int wv    = threadIdx.x >> 5;
  const int qt    = blockIdx.y * 8 + wv;
  const int chunk = blockIdx.x;
  const int g0    = chunk * CHUNK_G;
  const int gEnd  = min(g0 + CHUNK_G, GROUPS);

  // A fragments for this wave's query tile (K = 128 -> 4 chunks of 32)
  v16h a0 = *(const v16h*)(qn + ((size_t)(qt * 4 + 0) * 32 + lane) * 16);
  v16h a1 = *(const v16h*)(qn + ((size_t)(qt * 4 + 1) * 32 + lane) * 16);
  v16h a2 = *(const v16h*)(qn + ((size_t)(qt * 4 + 2) * 32 + lane) * 16);
  v16h a3 = *(const v16h*)(qn + ((size_t)(qt * 4 + 3) * 32 + lane) * 16);

  float tv[8][TOPK];
  int   ti[8][TOPK];
#pragma unroll
  for (int i = 0; i < 8; ++i)
#pragma unroll
    for (int j = 0; j < TOPK; ++j) { tv[i][j] = NEG_INF; ti[i][j] = 0; }

  for (int g = g0; g < gEnd; ++g) {
    const _Float16* bp = kn + (size_t)g * 2048 + lane * 16;
    v16h b0 = *(const v16h*)(bp);
    v16h b1 = *(const v16h*)(bp + 512);
    v16h b2 = *(const v16h*)(bp + 1024);
    v16h b3 = *(const v16h*)(bp + 1536);

    v8f c = {};
    c = __builtin_amdgcn_wmma_f32_16x16x32_f16(false, a0, false, b0,
                                               (short)0, c, false, false);
    c = __builtin_amdgcn_wmma_f32_16x16x32_f16(false, a1, false, b1,
                                               (short)0, c, false, false);
    c = __builtin_amdgcn_wmma_f32_16x16x32_f16(false, a2, false, b2,
                                               (short)0, c, false, false);
    c = __builtin_amdgcn_wmma_f32_16x16x32_f16(false, a3, false, b3,
                                               (short)0, c, false, false);

    const int key = g * 16 + (lane & 15);   // column index (same both halves)
    const float dec = decay[key];

    float s[8];
    bool improve = false;
#pragma unroll
    for (int i = 0; i < 8; ++i) {
      s[i] = c[i] * dec;
      improve = improve || (s[i] > tv[i][TOPK - 1]);
    }
    if (__builtin_amdgcn_ballot_w32(improve) != 0u) {
#pragma unroll
      for (int i = 0; i < 8; ++i) {
        float v = s[i]; int id = key;
#pragma unroll
        for (int j = 0; j < TOPK; ++j) {
          bool gt = v > tv[i][j];
          float ov = tv[i][j]; int oi = ti[i][j];
          tv[i][j] = gt ? v : ov;  ti[i][j] = gt ? id : oi;
          v = gt ? ov : v;         id = gt ? oi : id;
        }
      }
    }
  }

  // Two-phase cross-lane merge (16 lanes x top5 -> top5 per query row)
  __shared__ float lv[4][16][16][TOPK];
  __shared__ int   li[4][16][16][TOPK];
  for (int phase = 0; phase < 2; ++phase) {
    __syncthreads();
    if ((wv >> 2) == phase) {
      const int w2 = wv & 3, cl = lane & 15;
#pragma unroll
      for (int i = 0; i < 8; ++i) {
        int row = (lane < 16) ? i : (i + 8);
#pragma unroll
        for (int j = 0; j < TOPK; ++j) {
          lv[w2][row][cl][j] = tv[i][j];
          li[w2][row][cl][j] = ti[i][j];
        }
      }
    }
    __syncthreads();
    if (threadIdx.x < 64) {
      const int w2 = threadIdx.x >> 4, row = threadIdx.x & 15;
      const int wvG = phase * 4 + w2;
      const int q = blockIdx.y * 128 + wvG * 16 + row;
      float* V = &lv[w2][row][0][0];
      int*   I = &li[w2][row][0][0];
      for (int k = 0; k < TOPK; ++k) {
        float best = NEG_INF; int bid = 0, bpos = 0;
        for (int t = 0; t < 16 * TOPK; ++t)
          if (V[t] > best) { best = V[t]; bid = I[t]; bpos = t; }
        V[bpos] = NEG_INF;
        pv[((size_t)q * NCHUNK + chunk) * TOPK + k] = best;
        pi[((size_t)q * NCHUNK + chunk) * TOPK + k] = bid;
      }
    }
  }
}

// ---------------------------------------------------------------------------
// Kernel 5: final merge across chunks + gather values + write outputs
// out = [retrieved 215040][top_sims 1280][valid_mask 1280][top_idx 1280]
// ---------------------------------------------------------------------------
__global__ void final_kernel(const float* __restrict__ pv,
                             const int* __restrict__ pi,
                             const float* __restrict__ values,
                             float* __restrict__ out) {
  const int q = blockIdx.x;
  const int NC = NCHUNK * TOPK;   // 490
  __shared__ float lvv[NCHUNK * TOPK];
  __shared__ int   lii[NCHUNK * TOPK];
  __shared__ float bv[TOPK];
  __shared__ int   bi[TOPK];
  for (int t = threadIdx.x; t < NC; t += 64) {
    lvv[t] = pv[(size_t)q * NC + t];
    lii[t] = pi[(size_t)q * NC + t];
  }
  __syncthreads();
  if (threadIdx.x == 0) {
    for (int k = 0; k < TOPK; ++k) {
      float best = NEG_INF; int bid = 0, bpos = 0;
      for (int t = 0; t < NC; ++t)
        if (lvv[t] > best) { best = lvv[t]; bid = lii[t]; bpos = t; }
      lvv[bpos] = NEG_INF;
      bv[k] = best; bi[k] = bid;
    }
  }
  __syncthreads();
  float* retr  = out;
  float* osim  = out + (size_t)BATCH * TOPK * VROW;
  float* omask = osim + (size_t)BATCH * TOPK;
  float* oidx  = omask + (size_t)BATCH * TOPK;
  for (int t = threadIdx.x; t < TOPK * VROW; t += 64) {
    int k = t / VROW, r = t % VROW;
    retr[(size_t)q * TOPK * VROW + t] = values[(size_t)bi[k] * VROW + r];
  }
  if (threadIdx.x < TOPK) {
    int k = threadIdx.x;
    osim[q * TOPK + k]  = bv[k];
    omask[q * TOPK + k] = (bv[k] >= 0.0f) ? 1.0f : 0.0f;
    oidx[q * TOPK + k]  = (float)bi[k];
  }
}

// ---------------------------------------------------------------------------
extern "C" void kernel_launch(void* const* d_in, const int* in_sizes, int n_in,
                              void* d_out, int out_size, void* d_ws, size_t ws_size,
                              hipStream_t stream) {
  const float* query  = (const float*)d_in[0];
  const float* keys   = (const float*)d_in[1];
  const float* values = (const float*)d_in[2];
  const int*   ts     = (const int*)d_in[3];
  const int*   gstep  = (const int*)d_in[4];
  // d_in[5] (top_k) fixed at 5 by reference setup.

  char* w = (char*)d_ws;
  // ws layout (bytes):
  //   kn    : 6250*2048 f16 = 25,600,000
  //   qn    : 16*4*32*16 f16 =     65,536
  //   decay : 100000 f32     =    400,000
  //   pv/pi : 256*98*5 each  =    501,760 x2
  _Float16* ws_kn    = (_Float16*)(w);
  _Float16* ws_qn    = (_Float16*)(w + 25600000);
  float*    ws_decay = (float*)   (w + 25665536);
  float*    ws_pv    = (float*)   (w + 26065664);
  int*      ws_pi    = (int*)     (w + 26567424);

  prep_q_kernel<<<16, 64, 0, stream>>>(query, ws_qn);
  prep_k_kernel<<<GROUPS, 64, 0, stream>>>(keys, ws_kn);
  decay_kernel<<<(CAP + 255) / 256, 256, 0, stream>>>(ts, gstep, ws_decay);
  sim_topk_kernel<<<dim3(NCHUNK, 2), 256, 0, stream>>>(ws_kn, ws_qn, ws_decay,
                                                       ws_pv, ws_pi);
  final_kernel<<<BATCH, 64, 0, stream>>>(ws_pv, ws_pi, values, (float*)d_out);
}
